// MultiHeadLatentAttention_63720134804069
// MI455X (gfx1250) — matrix operand order repaired and checked by reading, not compiled, not run on hardware
//
#include <hip/hip_runtime.h>
#include <hip/hip_bf16.h>
#include <stdint.h>

// ---------------- problem constants ----------------
#define S_LEN   2048
#define DMODEL  2048
#define NHEADS  16
#define DHEAD   128
#define DC_DIM  512
#define DQC_DIM 1024

typedef __bf16 bf16_t;
typedef __attribute__((ext_vector_type(16))) __bf16 v16bf;
typedef __attribute__((ext_vector_type(8)))  __bf16 v8bf;
typedef __attribute__((ext_vector_type(8)))  float  v8f;
typedef __attribute__((ext_vector_type(4)))  unsigned int u32x4;
typedef __attribute__((ext_vector_type(8)))  int i32x8;
typedef __attribute__((ext_vector_type(4)))  int i32x4;

#if defined(__has_builtin)
#if __has_builtin(__builtin_amdgcn_tensor_load_to_lds)
#define HAVE_TDM 1
#endif
#endif

// ---------------- bf16 helpers (bit-exact RNE, no reliance on __bf16 casts) ----
__device__ __forceinline__ bf16_t f2bf(float f) {
  union { float f; unsigned u; } a; a.f = f;
  unsigned r = (a.u + 0x7FFFu + ((a.u >> 16) & 1u)) >> 16;
  union { unsigned short s; bf16_t b; } o; o.s = (unsigned short)r; return o.b;
}
__device__ __forceinline__ float bf2f(bf16_t b) {
  union { unsigned short s; bf16_t b; } i; i.b = b;
  union { unsigned u; float f; } a; a.u = ((unsigned)i.s) << 16;
  return a.f;
}

__device__ __forceinline__ v8f wmma_bf16(v16bf a, v16bf b, v8f c) {
  // (neg_a, A, neg_b, B, c_mod, C, reuse_a, reuse_b)
  return __builtin_amdgcn_wmma_f32_16x16x32_bf16(false, a, false, b, (short)0, c, false, false);
}

// A fragment: 16x32 bf16 tile, rows m0..m0+15 / cols k0..k0+31 of row-major [*, lda].
// A layout (matches B): lanes 0-15 hold K=0..15; lanes 16-31 hold K=16..31.
__device__ __forceinline__ v16bf load_a_frag(const bf16_t* A, int lda, int m0, int k0, int lane) {
  int hf = lane >> 4, r = lane & 15;
  const bf16_t* p = A + (size_t)(m0 + r) * lda + (size_t)(k0 + hf * 16);
  union { v16bf v; v8bf h[2]; } u;
  u.h[0] = *(const v8bf*)(p);
  u.h[1] = *(const v8bf*)(p + 8);
  return u.v;
}
// B fragment: 32x16 bf16 (KxN). Column n of B == row (n0+n) of the row-major [*, ldb] source.
// lanes 0-15: K=0..15, lanes 16-31: K=16..31.
__device__ __forceinline__ v16bf load_b_frag(const bf16_t* Bm, int ldb, int n0, int k0, int lane) {
  int hf = lane >> 4, n = lane & 15;
  const bf16_t* p = Bm + (size_t)(n0 + n) * ldb + (size_t)(k0 + hf * 16);
  return *(const v16bf*)(p);
}

// ---------------- Tensor Data Mover: 2D tile (tile_k x tile_rows) global -> LDS ----
// D# per CDNA5 ISA ch.8: group0 = {count/lds_addr/global_addr/type=2},
// group1 = {data_size=2B, tensor dims, tile dims, dim0 stride}. Groups 2/3 zero (<=2D).
__device__ __forceinline__ void tdm_load_tile_2d(unsigned lds_addr, const void* gptr,
                                                 unsigned tile_k, unsigned tile_rows,
                                                 unsigned row_stride_elems) {
#if defined(HAVE_TDM)
  unsigned long long ga = (unsigned long long)(uintptr_t)gptr;
  u32x4 g0;
  g0[0] = 1u;                                                 // count=1, user descriptor
  g0[1] = lds_addr;                                           // LDS byte address
  g0[2] = (unsigned)ga;                                       // global_addr[31:0]
  g0[3] = (unsigned)((ga >> 32) & 0x01FFFFFFu) | (2u << 30);  // global_addr[56:32], type=2
  i32x8 g1;
  g1[0] = (int)(1u << 16);                                    // wg_mask=0, data_size=1 (2B)
  g1[1] = (int)((tile_k & 0xFFFFu) << 16);                    // tensor_dim0[15:0] @ [63:48]
  g1[2] = (int)((tile_k >> 16) | ((tile_rows & 0xFFFFu) << 16)); // td0 hi / td1 lo
  g1[3] = (int)((tile_rows >> 16) | ((tile_k & 0xFFFFu) << 16)); // td1 hi / tile_dim0
  g1[4] = (int)(tile_rows & 0xFFFFu);                         // tile_dim1 (tile_dim2=0)
  g1[5] = (int)row_stride_elems;                              // tensor_dim0_stride[31:0]
  g1[6] = 0;                                                  // stride0 hi / stride1 lo
  g1[7] = 0;                                                  // stride1 hi
  i32x4 gz = (i32x4)0;
#if __clang_major__ >= 23
  i32x8 gz8 = (i32x8)0;
  __builtin_amdgcn_tensor_load_to_lds(g0, g1, gz, gz, gz8, 0);
#else
  __builtin_amdgcn_tensor_load_to_lds(g0, g1, gz, gz, 0);
#endif
#else
  (void)lds_addr; (void)gptr; (void)tile_k; (void)tile_rows; (void)row_stride_elems;
#endif
}

// ---------------- fp32 -> bf16 conversion ----------------
__global__ void cvt_f32_bf16(const float* __restrict__ x, bf16_t* __restrict__ y, size_t n) {
  size_t i = (size_t)blockIdx.x * blockDim.x + threadIdx.x;
  if (i < n) y[i] = f2bf(x[i]);
}

// ---------------- WMMA GEMM: C[M,N] = A[M,K] * W[N,K]^T + bias[N] -----------------
// 256 threads = 8 waves; block tile 128x128; wave tile 32x64 (2x4 wmma accumulators).
// TDM double-buffered LDS staging: BK=64, A/W tiles 16KB each, 2 buffers = 64KB LDS.
template <bool F32OUT>
__global__ __launch_bounds__(256)
void gemm_wmma_bf16(const bf16_t* __restrict__ A, const bf16_t* __restrict__ W,
                    const float* __restrict__ bias, void* __restrict__ Cout,
                    int M, int N, int K) {
#if defined(HAVE_TDM)
  __shared__ __attribute__((aligned(64))) bf16_t Atile[2][128][64];
  __shared__ __attribute__((aligned(64))) bf16_t Wtile[2][128][64];
#endif
  const int lane = threadIdx.x & 31;
  const int wave = threadIdx.x >> 5;       // 0..7
  const int wm = wave & 3;                 // 4 wave-rows (32 rows each)
  const int wn = wave >> 2;                // 2 wave-cols (64 cols each)
  const int mblk = blockIdx.y * 128;
  const int nblk = blockIdx.x * 128;

  v8f acc[2][4];
#pragma unroll
  for (int i = 0; i < 2; ++i)
#pragma unroll
    for (int j = 0; j < 4; ++j) acc[i][j] = (v8f)(0.0f);

#if defined(HAVE_TDM)
  auto issue_stage = [&](int buf, int k0) {
    tdm_load_tile_2d((unsigned)(uintptr_t)(void*)&Atile[buf][0][0],
                     A + (size_t)mblk * K + k0, 64u, 128u, (unsigned)K);
    tdm_load_tile_2d((unsigned)(uintptr_t)(void*)&Wtile[buf][0][0],
                     W + (size_t)nblk * K + k0, 64u, 128u, (unsigned)K);
  };
  auto compute_stage = [&](int buf) {
#pragma unroll
    for (int ks = 0; ks < 64; ks += 32) {
      v16bf a0 = load_a_frag(&Atile[buf][0][0], 64, wm * 32,      ks, lane);
      v16bf a1 = load_a_frag(&Atile[buf][0][0], 64, wm * 32 + 16, ks, lane);
      v16bf bfr[4];
#pragma unroll
      for (int j = 0; j < 4; ++j)
        bfr[j] = load_b_frag(&Wtile[buf][0][0], 64, wn * 64 + j * 16, ks, lane);
#pragma unroll
      for (int j = 0; j < 4; ++j) {
        acc[0][j] = wmma_bf16(a0, bfr[j], acc[0][j]);
        acc[1][j] = wmma_bf16(a1, bfr[j], acc[1][j]);
      }
    }
  };

  if (wave == 0) issue_stage(0, 0);
  int buf = 0;
  int k0 = 0;
  for (; k0 + 64 < K; k0 += 64, buf ^= 1) {
    if (wave == 0) {
      issue_stage(buf ^ 1, k0 + 64);                    // prefetch next stage via TDM
      __builtin_amdgcn_s_wait_tensorcnt((short)2);      // current stage's 2 DMAs done
    }
    __syncthreads();
    compute_stage(buf);
    __syncthreads();                                    // all readers done before reuse
  }
  if (wave == 0) __builtin_amdgcn_s_wait_tensorcnt((short)0);
  __syncthreads();
  compute_stage(buf);
#else
  // Fallback: direct-from-global fragments.
  for (int kk = 0; kk < K; kk += 32) {
    v16bf a0 = load_a_frag(A, K, mblk + wm * 32,      kk, lane);
    v16bf a1 = load_a_frag(A, K, mblk + wm * 32 + 16, kk, lane);
    v16bf bfr[4];
#pragma unroll
    for (int j = 0; j < 4; ++j) bfr[j] = load_b_frag(W, K, nblk + wn * 64 + j * 16, kk, lane);
#pragma unroll
    for (int j = 0; j < 4; ++j) {
      acc[0][j] = wmma_bf16(a0, bfr[j], acc[0][j]);
      acc[1][j] = wmma_bf16(a1, bfr[j], acc[1][j]);
    }
  }
#endif

  const int hf = lane >> 4, c = lane & 15;
#pragma unroll
  for (int i2 = 0; i2 < 2; ++i2)
#pragma unroll
    for (int j = 0; j < 4; ++j)
#pragma unroll
      for (int i = 0; i < 8; ++i) {
        int row = mblk + wm * 32 + i2 * 16 + i + hf * 8; // C layout: VGPR i -> M=i+8*(lane/16)
        int col = nblk + wn * 64 + j * 16 + c;
        float v = acc[i2][j][i] + bias[col];
        if (F32OUT) ((float*)Cout)[(size_t)row * N + col] = v;
        else        ((bf16_t*)Cout)[(size_t)row * N + col] = f2bf(v);
      }
}

// ---------------- pack K: K[h][s][0:128]=k_C, K[h][s][128:256]=rope(k_R) ----------------
__global__ void pack_k_kernel(const bf16_t* __restrict__ kC, const bf16_t* __restrict__ kR,
                              bf16_t* __restrict__ Kb) {
  int idx = blockIdx.x * blockDim.x + threadIdx.x;   // S*H*64 pair slots
  if (idx >= S_LEN * NHEADS * 64) return;
  int i = idx & 63;               // pair index within head dim (dh/2)
  int h = (idx >> 6) & (NHEADS - 1);
  int s = idx >> 10;
  float freq = __expf(-(float)i * (9.210340371976184f / 64.0f));  // 10000^(-i/64)
  float th = (float)s * freq;
  float cs = __cosf(th), sn = __sinf(th);
  size_t src = (size_t)s * DMODEL + (size_t)h * DHEAD + 2 * i;
  float e = bf2f(kR[src]), o = bf2f(kR[src + 1]);
  size_t dst = ((size_t)h * S_LEN + s) * 256;
  Kb[dst + 2 * i]           = kC[src];
  Kb[dst + 2 * i + 1]       = kC[src + 1];
  Kb[dst + 128 + 2 * i]     = f2bf(e * cs - o * sn);
  Kb[dst + 128 + 2 * i + 1] = f2bf(e * sn + o * cs);
}

// ---------------- pack Q: Q[h][s][0:128]=q_C, Q[h][s][128:256]=rope(q_C) ----------------
__global__ void pack_q_kernel(const bf16_t* __restrict__ qC, bf16_t* __restrict__ Qb) {
  int idx = blockIdx.x * blockDim.x + threadIdx.x;
  if (idx >= S_LEN * NHEADS * 64) return;
  int i = idx & 63;
  int h = (idx >> 6) & (NHEADS - 1);
  int s = idx >> 10;
  float freq = __expf(-(float)i * (9.210340371976184f / 64.0f));
  float th = (float)s * freq;
  float cs = __cosf(th), sn = __sinf(th);
  size_t src = (size_t)s * DMODEL + (size_t)h * DHEAD + 2 * i;
  float e = bf2f(qC[src]), o = bf2f(qC[src + 1]);
  size_t dst = ((size_t)h * S_LEN + s) * 256;
  Qb[dst + 2 * i]           = qC[src];
  Qb[dst + 2 * i + 1]       = qC[src + 1];
  Qb[dst + 128 + 2 * i]     = f2bf(e * cs - o * sn);
  Qb[dst + 128 + 2 * i + 1] = f2bf(e * sn + o * cs);
}

// ---------------- pack V^T: VT[h][dh][s] = v[s][h*128+dh] ----------------
__global__ void pack_vt_kernel(const bf16_t* __restrict__ v, bf16_t* __restrict__ VT) {
  int idx = blockIdx.x * blockDim.x + threadIdx.x;   // H*128*S
  if (idx >= NHEADS * DHEAD * S_LEN) return;
  int s = idx & (S_LEN - 1);
  int d = (idx >> 11) & (DHEAD - 1);
  int h = idx >> 18;
  VT[idx] = v[(size_t)s * DMODEL + (size_t)h * DHEAD + d];
}

// ---------------- causal flash attention ----------------
// grid: (S/64, H); block 128 = 4 waves; each wave owns 16 query rows.
// Q,K: [H][S][256] bf16; VT: [H][128][S] bf16; O: [S][2048] bf16 (head-interleaved).
__global__ __launch_bounds__(128)
void mla_attention_kernel(const bf16_t* __restrict__ Qb, const bf16_t* __restrict__ Kb,
                          const bf16_t* __restrict__ VT, bf16_t* __restrict__ Obf) {
  __shared__ __attribute__((aligned(64))) bf16_t Pst[4][16][32];  // per-wave P staging
  const int lane = threadIdx.x & 31;
  const int wave = threadIdx.x >> 5;
  const int h  = blockIdx.y;
  const int q0 = blockIdx.x * 64 + wave * 16;
  const int hf = lane >> 4, c16 = lane & 15;

  const bf16_t* Qh = Qb + (size_t)h * S_LEN * 256;
  const bf16_t* Kh = Kb + (size_t)h * S_LEN * 256;
  const bf16_t* Vh = VT + (size_t)h * DHEAD * S_LEN;

  // Q fragments resident in registers: 16 rows x 256 -> 8 chunks of K=32
  v16bf qf[8];
#pragma unroll
  for (int cI = 0; cI < 8; ++cI) qf[cI] = load_a_frag(Qh, 256, q0, cI * 32, lane);

  v8f o[8];
#pragma unroll
  for (int t = 0; t < 8; ++t) o[t] = (v8f)(0.0f);
  float mrow[8], lrow[8];
#pragma unroll
  for (int i = 0; i < 8; ++i) { mrow[i] = -__builtin_inff(); lrow[i] = 0.0f; }

  const float scale = 0.0625f;           // 1/sqrt(2*DH) = 1/16
  const int qmax = q0 + 15;

  for (int j0 = 0; j0 <= qmax; j0 += 32) {
    // speculative prefetch of the next key block (global_prefetch_b8)
    if (j0 + 48 < S_LEN)
      __builtin_prefetch((const void*)(Kh + (size_t)(j0 + 32 + c16) * 256), 0, 3);

    // ---- scores: S = Q(16x256) * K_blk(256x32)^T as two 16x16 wmma accumulators
    v8f s0 = (v8f)(0.0f), s1 = (v8f)(0.0f);
#pragma unroll
    for (int cI = 0; cI < 8; ++cI) {
      v16bf k0f = load_b_frag(Kh, 256, j0,      cI * 32, lane);
      v16bf k1f = load_b_frag(Kh, 256, j0 + 16, cI * 32, lane);
      s0 = wmma_bf16(qf[cI], k0f, s0);
      s1 = wmma_bf16(qf[cI], k1f, s1);
    }
    // ---- online softmax (rows live across 16-lane halves of the wave)
    float p0[8], p1[8], alpha[8];
#pragma unroll
    for (int i = 0; i < 8; ++i) {
      int row = q0 + i + hf * 8;
      float v0 = s0[i] * scale, v1 = s1[i] * scale;
      if (j0 + c16 > row)      v0 = -__builtin_inff();   // causal mask
      if (j0 + 16 + c16 > row) v1 = -__builtin_inff();
      float mx = fmaxf(v0, v1);
#pragma unroll
      for (int off = 1; off < 16; off <<= 1) mx = fmaxf(mx, __shfl_xor(mx, off, 32));
      float mnew = fmaxf(mrow[i], mx);
      alpha[i] = __expf(mrow[i] - mnew);
      p0[i] = __expf(v0 - mnew);
      p1[i] = __expf(v1 - mnew);
      float rs = p0[i] + p1[i];
#pragma unroll
      for (int off = 1; off < 16; off <<= 1) rs += __shfl_xor(rs, off, 32);
      lrow[i] = lrow[i] * alpha[i] + rs;
      mrow[i] = mnew;
    }
#pragma unroll
    for (int t = 0; t < 8; ++t)
#pragma unroll
      for (int i = 0; i < 8; ++i) o[t][i] *= alpha[i];

    // ---- stage P (16x32) through per-wave LDS to reshape C-layout -> A-layout
#pragma unroll
    for (int i = 0; i < 8; ++i) {
      Pst[wave][i + hf * 8][c16]      = f2bf(p0[i]);
      Pst[wave][i + hf * 8][16 + c16] = f2bf(p1[i]);
    }
    // LDS ops from one wave complete in order; just wait for the stores.
    asm volatile("s_wait_dscnt 0" ::: "memory");
    __builtin_amdgcn_wave_barrier();
    v16bf pf;
    {
      const bf16_t* pp = &Pst[wave][c16][hf * 16];
      union { v16bf v; v8bf hh[2]; } u;
      u.hh[0] = *(const v8bf*)(pp);
      u.hh[1] = *(const v8bf*)(pp + 8);
      pf = u.v;
    }
    // ---- O += P(16x32) * V_blk(32x128): VT rows are contiguous in key dim
#pragma unroll
    for (int t = 0; t < 8; ++t) {
      const bf16_t* vp = Vh + (size_t)(t * 16 + c16) * S_LEN + (size_t)(j0 + hf * 16);
      v16bf vf = *(const v16bf*)(vp);
      o[t] = wmma_bf16(pf, vf, o[t]);
    }
    __builtin_amdgcn_wave_barrier();     // keep next iter's P stores behind these reads
  }

  // ---- normalize and store: O[s][h*128 + dh]
#pragma unroll
  for (int t = 0; t < 8; ++t)
#pragma unroll
    for (int i = 0; i < 8; ++i) {
      int row = q0 + i + hf * 8;
      float v = o[t][i] / lrow[i];
      Obf[(size_t)row * DMODEL + (size_t)h * DHEAD + t * 16 + c16] = f2bf(v);
    }
}

// ---------------- host-side orchestration ----------------
extern "C" void kernel_launch(void* const* d_in, const int* in_sizes, int n_in,
                              void* d_out, int out_size, void* d_ws, size_t ws_size,
                              hipStream_t stream) {
  (void)in_sizes; (void)n_in; (void)out_size; (void)ws_size;

  const float* h_f    = (const float*)d_in[0];
  const float* wdkv_f = (const float*)d_in[1];
  const float* bdkv   = (const float*)d_in[2];
  const float* wuk_f  = (const float*)d_in[3];
  const float* buk    = (const float*)d_in[4];
  const float* wuv_f  = (const float*)d_in[5];
  const float* buv    = (const float*)d_in[6];
  const float* wdq_f  = (const float*)d_in[7];
  const float* bdq    = (const float*)d_in[8];
  const float* wuq_f  = (const float*)d_in[9];
  const float* buq    = (const float*)d_in[10];
  const float* wkr_f  = (const float*)d_in[11];
  const float* bkr    = (const float*)d_in[12];
  const float* wo_f   = (const float*)d_in[13];
  const float* bo     = (const float*)d_in[14];
  float* out = (float*)d_out;

  // workspace carve-up (bf16 buffers, 256B aligned)
  char* wp = (char*)d_ws;
  auto take = [&](size_t elems) -> bf16_t* {
    bf16_t* r = (bf16_t*)wp;
    wp += ((elems * sizeof(bf16_t)) + 255) & ~(size_t)255;
    return r;
  };
  bf16_t* h_bf   = take((size_t)S_LEN * DMODEL);
  bf16_t* wdkv   = take((size_t)DC_DIM * DMODEL);
  bf16_t* wuk    = take((size_t)DMODEL * DC_DIM);
  bf16_t* wuv    = take((size_t)DMODEL * DC_DIM);
  bf16_t* wdq    = take((size_t)DQC_DIM * DMODEL);
  bf16_t* wuq    = take((size_t)DMODEL * DQC_DIM);
  bf16_t* wkr    = take((size_t)DMODEL * DMODEL);
  bf16_t* wo     = take((size_t)DMODEL * DMODEL);
  bf16_t* ckv    = take((size_t)S_LEN * DC_DIM);
  bf16_t* cq     = take((size_t)S_LEN * DQC_DIM);
  bf16_t* kC     = take((size_t)S_LEN * DMODEL);
  bf16_t* kR     = take((size_t)S_LEN * DMODEL);
  bf16_t* qC     = take((size_t)S_LEN * DMODEL);
  bf16_t* vC     = take((size_t)S_LEN * DMODEL);
  bf16_t* Kbuf   = take((size_t)NHEADS * S_LEN * 256);
  bf16_t* Qbuf   = take((size_t)NHEADS * S_LEN * 256);
  bf16_t* VTbuf  = take((size_t)NHEADS * DHEAD * S_LEN);
  bf16_t* Obf    = take((size_t)S_LEN * DMODEL);

  const int TPB = 256;
  auto cvt = [&](const float* src, bf16_t* dst, size_t n) {
    cvt_f32_bf16<<<dim3((unsigned)((n + TPB - 1) / TPB)), dim3(TPB), 0, stream>>>(src, dst, n);
  };
  cvt(h_f,    h_bf, (size_t)S_LEN * DMODEL);
  cvt(wdkv_f, wdkv, (size_t)DC_DIM * DMODEL);
  cvt(wuk_f,  wuk,  (size_t)DMODEL * DC_DIM);
  cvt(wuv_f,  wuv,  (size_t)DMODEL * DC_DIM);
  cvt(wdq_f,  wdq,  (size_t)DQC_DIM * DMODEL);
  cvt(wuq_f,  wuq,  (size_t)DMODEL * DQC_DIM);
  cvt(wkr_f,  wkr,  (size_t)DMODEL * DMODEL);
  cvt(wo_f,   wo,   (size_t)DMODEL * DMODEL);

  auto gemm_bf = [&](const bf16_t* A, const bf16_t* W, const float* bias, bf16_t* C,
                     int M, int N, int K) {
    gemm_wmma_bf16<false><<<dim3(N / 128, M / 128), dim3(256), 0, stream>>>(A, W, bias,
                                                                            (void*)C, M, N, K);
  };

  // latent KV path
  gemm_bf(h_bf, wdkv, bdkv, ckv, S_LEN, DC_DIM, DMODEL);     // c_kv  [S,512]
  gemm_bf(ckv,  wuk,  buk,  kC,  S_LEN, DMODEL, DC_DIM);     // k_C   [S,2048]
  gemm_bf(ckv,  wuv,  buv,  vC,  S_LEN, DMODEL, DC_DIM);     // v_C   [S,2048]
  // RoPE K path
  gemm_bf(h_bf, wkr,  bkr,  kR,  S_LEN, DMODEL, DMODEL);     // k_R   [S,2048]
  // latent Q path
  gemm_bf(h_bf, wdq,  bdq,  cq,  S_LEN, DQC_DIM, DMODEL);    // c_q   [S,1024]
  gemm_bf(cq,   wuq,  buq,  qC,  S_LEN, DMODEL, DQC_DIM);    // q_C   [S,2048]

  // pack K/Q (with RoPE) and V^T
  {
    int npair = S_LEN * NHEADS * 64;
    pack_k_kernel<<<dim3((npair + TPB - 1) / TPB), dim3(TPB), 0, stream>>>(kC, kR, Kbuf);
    pack_q_kernel<<<dim3((npair + TPB - 1) / TPB), dim3(TPB), 0, stream>>>(qC, Qbuf);
    int nvt = NHEADS * DHEAD * S_LEN;
    pack_vt_kernel<<<dim3((nvt + TPB - 1) / TPB), dim3(TPB), 0, stream>>>(vC, VTbuf);
  }

  // causal flash attention
  mla_attention_kernel<<<dim3(S_LEN / 64, NHEADS), dim3(128), 0, stream>>>(Qbuf, Kbuf, VTbuf, Obf);

  // output projection (f32 result)
  gemm_wmma_bf16<true><<<dim3(DMODEL / 128, S_LEN / 128), dim3(256), 0, stream>>>(
      Obf, wo, bo, (void*)out, S_LEN, DMODEL, DMODEL);
}
